// Attn_10849087390349
// MI455X (gfx1250) — matrix operand order repaired and checked by reading, not compiled, not run on hardware
//
#include <hip/hip_runtime.h>
#include <cstdint>

// ---------------------------------------------------------------------------
// MHA forward for B=2, L=2048, D=1024, H=16, DK=64 on gfx1250 (CDNA5, wave32)
// Pipeline: fp32->bf16 convert, QKV GEMM (WMMA bf16, async-LDS double-buffer),
// flash attention (WMMA bf16, fp32 online softmax, async K staging),
// output GEMM (WMMA bf16 -> fp32).
// ---------------------------------------------------------------------------

typedef __bf16 bf16;
typedef __attribute__((ext_vector_type(16))) __bf16 v16bf;
typedef __attribute__((ext_vector_type(8)))  __bf16 v8bf;
typedef __attribute__((ext_vector_type(4)))  __bf16 v4bf;
typedef __attribute__((ext_vector_type(8)))  float  v8f;

__device__ __forceinline__ bf16 f2bf(float f) {
  unsigned u = __builtin_bit_cast(unsigned, f);
  unsigned r = u + 0x7FFFu + ((u >> 16) & 1u);   // round-to-nearest-even
  unsigned short h = (unsigned short)(r >> 16);
  return __builtin_bit_cast(bf16, h);
}

__device__ __forceinline__ v8f vzero8() {
  v8f z;
#pragma unroll
  for (int i = 0; i < 8; ++i) z[i] = 0.0f;
  return z;
}

// LDS byte offset of a __shared__ object: flat addr low 32 bits.
__device__ __forceinline__ unsigned lds_off(const void* p) {
  return (unsigned)(uintptr_t)p;
}

// Async copy 16 bytes global -> LDS (GLOBAL_LOAD_ASYNC_TO_LDS_B128, ASYNCcnt).
__device__ __forceinline__ void async_b128(unsigned lds, const void* gptr) {
  asm volatile("global_load_async_to_lds_b128 %0, %1, off"
               :: "v"(lds), "v"(gptr) : "memory");
}

__device__ __forceinline__ void wait_async0() {
  asm volatile("s_wait_asynccnt 0x0" ::: "memory");
}

// A fragment (16x32 bf16, row-major source, leading dim lda in elements).
// ISA 7.12.2: lane half h, row m=lane&15 holds K = {h*8..h*8+7, 16+h*8..16+h*8+7}.
__device__ __forceinline__ v16bf load_a_frag(const bf16* base, int lda) {
  const int lane = threadIdx.x & 31;
  const int m  = lane & 15;
  const int hh = lane >> 4;
  const bf16* p = base + m * lda + hh * 8;
  v8bf lo = *(const v8bf*)(p);
  v8bf hi = *(const v8bf*)(p + 16);
  return __builtin_shufflevector(lo, hi, 0,1,2,3,4,5,6,7,8,9,10,11,12,13,14,15);
}

// B fragment (32x16 bf16) from an [N][K]-layout tile (leading dim ldb).
// Lane n=lane&15 holds column n, K = {h*8..h*8+7, 16+h*8..16+h*8+7}, h=lane>>4.
__device__ __forceinline__ v16bf load_b_frag(const bf16* baseT, int ldb) {
  const int lane = threadIdx.x & 31;
  const int n  = lane & 15;
  const int kb = (lane >> 4) * 8;
  const bf16* p = baseT + n * ldb + kb;
  v8bf lo = *(const v8bf*)(p);
  v8bf hi = *(const v8bf*)(p + 16);
  return __builtin_shufflevector(lo, hi, 0,1,2,3,4,5,6,7,8,9,10,11,12,13,14,15);
}

// ---------------------------------------------------------------------------
// fp32 -> bf16 conversion (vectorized; n4 = element count / 4)
// ---------------------------------------------------------------------------
__global__ __launch_bounds__(256) void cvt_f32_bf16(const float* __restrict__ s,
                                                    bf16* __restrict__ d, int n4) {
  int i = blockIdx.x * 256 + threadIdx.x;
  if (i < n4) {
    float4 v = ((const float4*)s)[i];
    v4bf o;
    o[0] = f2bf(v.x); o[1] = f2bf(v.y); o[2] = f2bf(v.z); o[3] = f2bf(v.w);
    ((v4bf*)d)[i] = o;
  }
}

// ---------------------------------------------------------------------------
// GEMM: C[m][n] = sum_k A[m][k] * W[n][k]   (A row-major MxK, W row-major NxK)
// Block tile 128x128, BK=32, 8 waves, wave tile 32x64 (2x4 WMMA accums).
// Tiles staged with GLOBAL_LOAD_ASYNC_TO_LDS_B128, double-buffered, K-loop
// unrolled by 2 so buffer indices are compile-time constant:
// one s_wait_asynccnt + one barrier per K-step, no buffer-select VALU.
// mode 0: write fp32 C row-major (out projection).
// mode 1: scatter bf16 into Q/K/V [B,H,L,DK]; Q scaled by 1/sqrt(64).
// ---------------------------------------------------------------------------
#define GBK 32
#define GLD 40   // padded LDS row stride (elements); 80B rows, 16B aligned

__global__ __launch_bounds__(256) void gemm_wmma(
    const bf16* __restrict__ A, const bf16* __restrict__ W,
    int M, int N, int K,
    float* __restrict__ outF,
    bf16* __restrict__ Qo, bf16* __restrict__ Ko, bf16* __restrict__ Vo,
    int mode) {
  __shared__ __align__(16) bf16 At[2][128 * GLD];
  __shared__ __align__(16) bf16 Bt[2][128 * GLD];

  const int tid  = threadIdx.x;
  const int wave = tid >> 5;
  const int lane = tid & 31;
  const int wm   = wave & 3;   // 4 waves over M (32 rows each)
  const int wn   = wave >> 2;  // 2 waves over N (64 cols each)
  const int m0   = blockIdx.y * 128;
  const int n0   = blockIdx.x * 128;

  v8f acc[2][4];
#pragma unroll
  for (int i = 0; i < 2; ++i)
#pragma unroll
    for (int j = 0; j < 4; ++j) acc[i][j] = vzero8();

  const int lr = tid >> 1;          // 0..127
  const int lc = (tid & 1) * 16;    // 0 or 16

  // per-thread global sources (row base) and LDS destinations (both buffers)
  const bf16* sa = A + (size_t)(m0 + lr) * K + lc;
  const bf16* sb = W + (size_t)(n0 + lr) * K + lc;
  const unsigned la[2] = { lds_off(&At[0][lr * GLD + lc]),
                           lds_off(&At[1][lr * GLD + lc]) };
  const unsigned lb[2] = { lds_off(&Bt[0][lr * GLD + lc]),
                           lds_off(&Bt[1][lr * GLD + lc]) };

  auto stage = [&](int buf, int k0) {
    async_b128(la[buf],      sa + k0);
    async_b128(la[buf] + 16, sa + k0 + 8);
    async_b128(lb[buf],      sb + k0);
    async_b128(lb[buf] + 16, sb + k0 + 8);
  };

  // preload all fragments of a tile, then issue the 8 WMMAs (distinct regs ->
  // staggered partial dscnt waits instead of full drains between WMMA pairs)
  auto compute = [&](const bf16* at, const bf16* bt) {
    v16bf af[2], bfv[4];
#pragma unroll
    for (int mi = 0; mi < 2; ++mi)
      af[mi] = load_a_frag(at + (wm * 32 + mi * 16) * GLD, GLD);
#pragma unroll
    for (int ni = 0; ni < 4; ++ni)
      bfv[ni] = load_b_frag(bt + (wn * 64 + ni * 16) * GLD, GLD);
#pragma unroll
    for (int ni = 0; ni < 4; ++ni)
#pragma unroll
      for (int mi = 0; mi < 2; ++mi)
        acc[mi][ni] = __builtin_amdgcn_wmma_f32_16x16x32_bf16(
            false, af[mi], false, bfv[ni], (short)0, acc[mi][ni], false, false);
  };

  stage(0, 0);
  // K / GBK is even (K = 1024): unroll by 2 with static buffer indices.
  for (int k0 = 0; k0 + 2 * GBK <= K; k0 += 2 * GBK) {
    wait_async0();        // tile in buf0 fully written by this wave
    __syncthreads();      // ... and by all waves; buf1 readers (prev) done
    stage(1, k0 + GBK);
    compute(&At[0][0], &Bt[0][0]);

    wait_async0();
    __syncthreads();
    if (k0 + 2 * GBK < K) stage(0, k0 + 2 * GBK);
    compute(&At[1][0], &Bt[1][0]);
  }

  const int nl = lane & 15;
  const int hh = lane >> 4;
  if (mode == 0) {
#pragma unroll
    for (int mi = 0; mi < 2; ++mi)
#pragma unroll
      for (int ni = 0; ni < 4; ++ni)
#pragma unroll
        for (int r = 0; r < 8; ++r) {
          int m = m0 + wm * 32 + mi * 16 + hh * 8 + r;
          int n = n0 + wn * 64 + ni * 16 + nl;
          outF[(size_t)m * N + n] = acc[mi][ni][r];
        }
  } else {
#pragma unroll
    for (int mi = 0; mi < 2; ++mi)
#pragma unroll
      for (int ni = 0; ni < 4; ++ni)
#pragma unroll
        for (int r = 0; r < 8; ++r) {
          int m = m0 + wm * 32 + mi * 16 + hh * 8 + r;
          int e = n0 + wn * 64 + ni * 16 + nl;   // 0..3071
          int which = e >> 10;
          int h = (e >> 6) & 15;
          int dk = e & 63;
          int b = m >> 11;
          int l = m & 2047;
          float v = acc[mi][ni][r];
          bf16* dst;
          if (which == 0) { dst = Qo; v *= 0.125f; }     // fold 1/sqrt(DK)
          else if (which == 1) dst = Ko;
          else dst = Vo;
          dst[((size_t)(b * 16 + h) * 2048 + l) * 64 + dk] = f2bf(v);
        }
  }
}

// ---------------------------------------------------------------------------
// Flash attention: one block = 128 query rows of one (b,h); 8 waves x 16 rows.
// K tile (128x64) row-major in LDS (async-to-LDS staged), V tile transposed
// to [64][128] in LDS, P staged through per-wave LDS scratch for the
// C-layout -> A-layout reshape (legal without barrier: LDS ops are in-order).
// ---------------------------------------------------------------------------
#define KTP 72    // Kt row stride (elements)
#define VTP 136   // Vt row stride (elements)

__global__ __launch_bounds__(256) void attn_wmma(
    const bf16* __restrict__ Qg, const bf16* __restrict__ Kg,
    const bf16* __restrict__ Vg, bf16* __restrict__ ctx) {
  __shared__ __align__(16) bf16 Kt[128 * KTP];
  __shared__ __align__(16) bf16 Vt[64 * VTP];
  __shared__ __align__(16) bf16 Pl[8][16 * 128];

  const int tid  = threadIdx.x;
  const int wave = tid >> 5;
  const int lane = tid & 31;
  const int nl   = lane & 15;
  const int hh   = lane >> 4;
  const int bh   = blockIdx.y;              // b*16 + h
  const int q0   = blockIdx.x * 128;
  const size_t base = (size_t)bh * 2048 * 64;

  // Q fragments (16 rows x 64, pre-scaled) held in registers for whole kernel
  const bf16* qp = Qg + base + (size_t)(q0 + wave * 16) * 64;
  v16bf aq0 = load_a_frag(qp, 64);
  v16bf aq1 = load_a_frag(qp + 32, 64);

  v8f cacc[4];
#pragma unroll
  for (int d = 0; d < 4; ++d) cacc[d] = vzero8();
  float mrow[8], lsum[8];
#pragma unroll
  for (int r = 0; r < 8; ++r) { mrow[r] = -3.0e38f; lsum[r] = 0.0f; }

  const int lr = tid >> 1;        // 0..127
  const int lc = (tid & 1) * 32;  // 0 or 32

  const bf16* ksrow = Kg + base + (size_t)lr * 64 + lc;
  const bf16* vsrow = Vg + base + (size_t)lr * 64 + lc;
  unsigned lk = lds_off(&Kt[lr * KTP + lc]);

  for (int j0 = 0; j0 < 2048; j0 += 128) {
    // stage K tile row-major via async-to-LDS (4 x b128 per thread)
    const bf16* ks = ksrow + (size_t)j0 * 64;
#pragma unroll
    for (int c = 0; c < 4; ++c)
      async_b128(lk + c * 16, ks + c * 8);

    // stage V tile transposed through registers
    const bf16* vs = vsrow + (size_t)j0 * 64;
#pragma unroll
    for (int c = 0; c < 4; ++c) {
      v8bf vv = *(const v8bf*)(vs + c * 8);
#pragma unroll
      for (int e = 0; e < 8; ++e)
        Vt[(lc + c * 8 + e) * VTP + lr] = vv[e];
    }
    wait_async0();
    __syncthreads();

    // S = Q * K^T  (16 x 128), fp32 accum
    v8f s[8];
#pragma unroll
    for (int ni = 0; ni < 8; ++ni) {
      s[ni] = vzero8();
      v16bf b0 = load_b_frag(&Kt[(ni * 16) * KTP + 0], KTP);
      v16bf b1 = load_b_frag(&Kt[(ni * 16) * KTP + 32], KTP);
      s[ni] = __builtin_amdgcn_wmma_f32_16x16x32_bf16(
          false, aq0, false, b0, (short)0, s[ni], false, false);
      s[ni] = __builtin_amdgcn_wmma_f32_16x16x32_bf16(
          false, aq1, false, b1, (short)0, s[ni], false, false);
    }

    // online softmax: row max over 128 cols (8 accums + 16-lane shuffle tree)
    float alpha[8];
#pragma unroll
    for (int r = 0; r < 8; ++r) {
      float v = s[0][r];
#pragma unroll
      for (int ni = 1; ni < 8; ++ni) v = fmaxf(v, s[ni][r]);
      v = fmaxf(v, __shfl_xor(v, 1, 32));
      v = fmaxf(v, __shfl_xor(v, 2, 32));
      v = fmaxf(v, __shfl_xor(v, 4, 32));
      v = fmaxf(v, __shfl_xor(v, 8, 32));
      float mn = fmaxf(mrow[r], v);
      alpha[r] = __expf(mrow[r] - mn);
      mrow[r] = mn;
    }
#pragma unroll
    for (int d = 0; d < 4; ++d)
#pragma unroll
      for (int r = 0; r < 8; ++r) cacc[d][r] *= alpha[r];

    float rs[8];
#pragma unroll
    for (int r = 0; r < 8; ++r) rs[r] = 0.0f;
#pragma unroll
    for (int ni = 0; ni < 8; ++ni)
#pragma unroll
      for (int r = 0; r < 8; ++r) {
        float p = __expf(s[ni][r] - mrow[r]);
        s[ni][r] = p;
        rs[r] += p;
      }
#pragma unroll
    for (int r = 0; r < 8; ++r) {
      float v = rs[r];
      v += __shfl_xor(v, 1, 32);
      v += __shfl_xor(v, 2, 32);
      v += __shfl_xor(v, 4, 32);
      v += __shfl_xor(v, 8, 32);
      lsum[r] = lsum[r] * alpha[r] + v;
    }

    // P (C-layout regs) -> per-wave LDS scratch (row-major 16x128).
    bf16* pw = &Pl[wave][0];
#pragma unroll
    for (int ni = 0; ni < 8; ++ni)
#pragma unroll
      for (int r = 0; r < 8; ++r)
        pw[(hh * 8 + r) * 128 + ni * 16 + nl] = f2bf(s[ni][r]);

    // ctx += P * V   (16x128 @ 128x64), 4 K-steps of 32
#pragma unroll
    for (int kk = 0; kk < 4; ++kk) {
      v16bf ap = load_a_frag(pw + kk * 32, 128);
#pragma unroll
      for (int d = 0; d < 4; ++d) {
        v16bf bv = load_b_frag(&Vt[(d * 16) * VTP + kk * 32], VTP);
        cacc[d] = __builtin_amdgcn_wmma_f32_16x16x32_bf16(
            false, ap, false, bv, (short)0, cacc[d], false, false);
      }
    }
    __syncthreads();  // protect Kt/Vt before next tile staging
  }

  // normalize and store ctx as bf16 [B][L][H*DK]
  const int b = bh >> 4, h = bh & 15;
#pragma unroll
  for (int d = 0; d < 4; ++d)
#pragma unroll
    for (int r = 0; r < 8; ++r) {
      int qr = q0 + wave * 16 + hh * 8 + r;
      float v = cacc[d][r] / lsum[r];
      ctx[(size_t)(b * 2048 + qr) * 1024 + h * 64 + d * 16 + nl] = f2bf(v);
    }
}

// ---------------------------------------------------------------------------
extern "C" void kernel_launch(void* const* d_in, const int* in_sizes, int n_in,
                              void* d_out, int out_size, void* d_ws, size_t ws_size,
                              hipStream_t stream) {
  (void)in_sizes; (void)n_in; (void)out_size; (void)ws_size;
  const float* x     = (const float*)d_in[0];
  const float* w_qkv = (const float*)d_in[1];
  const float* w_o   = (const float*)d_in[2];
  float* out = (float*)d_out;

  const int D = 1024;
  const size_t M = 4096;  // B*L

  char* ws = (char*)d_ws;
  bf16* xb  = (bf16*)ws; ws += M * D * 2;          // x in bf16
  bf16* wqb = (bf16*)ws; ws += (size_t)3 * D * D * 2;
  bf16* wob = (bf16*)ws; ws += (size_t)D * D * 2;
  bf16* Qb  = (bf16*)ws; ws += M * D * 2;          // [B,H,L,DK], pre-scaled
  bf16* Kb  = (bf16*)ws; ws += M * D * 2;
  bf16* Vb  = (bf16*)ws; ws += M * D * 2;
  bf16* Cb  = (bf16*)ws; ws += M * D * 2;          // ctx [B,L,D]

  int n4;
  n4 = (int)(M * D / 4);
  cvt_f32_bf16<<<(n4 + 255) / 256, 256, 0, stream>>>(x, xb, n4);
  n4 = 3 * D * D / 4;
  cvt_f32_bf16<<<(n4 + 255) / 256, 256, 0, stream>>>(w_qkv, wqb, n4);
  n4 = D * D / 4;
  cvt_f32_bf16<<<(n4 + 255) / 256, 256, 0, stream>>>(w_o, wob, n4);

  // QKV projection: [4096,1024] x [3072,1024]^T -> Q/K/V scatter
  gemm_wmma<<<dim3(3 * D / 128, (int)(M / 128)), 256, 0, stream>>>(
      xb, wqb, (int)M, 3 * D, D, nullptr, Qb, Kb, Vb, 1);

  // attention: grid = (L/128 query tiles, B*H heads)
  attn_wmma<<<dim3(2048 / 128, 32), 256, 0, stream>>>(Qb, Kb, Vb, Cb);

  // output projection: [4096,1024] x [1024,1024]^T -> fp32 out
  gemm_wmma<<<dim3(D / 128, (int)(M / 128)), 256, 0, stream>>>(
      Cb, wob, (int)M, D, D, out, nullptr, nullptr, nullptr, 0);
}
